// AttnDecoderRNN_30159260352787
// MI455X (gfx1250) — compile-verified, matrix operand order checked by reading, not run on hardware
//
#include <hip/hip_runtime.h>
#include <hip/hip_bf16.h>
#include <math.h>

typedef __attribute__((ext_vector_type(16))) _Float16 v16h;
typedef __attribute__((ext_vector_type(8)))  float    v8f;

#define H_DIM 1024
#define V_DIM 50257
#define L_DIM 512

// ---------------------------------------------------------------------------
// K1: xcat = [ emb[x] | h0 ]   (2048 floats)
// ---------------------------------------------------------------------------
__global__ void k_build_xcat(const int* __restrict__ x,
                             const float* __restrict__ emb,
                             const float* __restrict__ hidden,
                             float* __restrict__ xcat) {
    int i = blockIdx.x * blockDim.x + threadIdx.x;
    if (i < H_DIM) {
        xcat[i] = emb[(size_t)x[0] * H_DIM + i];
    } else if (i < 2 * H_DIM) {
        xcat[i] = hidden[i - H_DIM];
    }
}

// ---------------------------------------------------------------------------
// Generic wave-per-row GEMV: y[r] = dot(W[r,:], x) + b[r]  (optional relu)
// W row-major, row stride == K.  K must be a multiple of 128.
// ---------------------------------------------------------------------------
__global__ void k_gemv_rows(const float* __restrict__ W,
                            const float* __restrict__ x,
                            const float* __restrict__ b,
                            float* __restrict__ y,
                            int rows, int K, int relu) {
    int gw   = (blockIdx.x * blockDim.x + threadIdx.x) >> 5;
    int lane = threadIdx.x & 31;
    if (gw >= rows) return;
    const float4* wr = (const float4*)(W + (size_t)gw * K);
    const float4* x4 = (const float4*)x;
    int K4 = K >> 2;
    float acc = 0.f;
    for (int k = lane; k < K4; k += 32) {
        float4 w = wr[k];
        float4 v = x4[k];
        acc = fmaf(w.x, v.x, fmaf(w.y, v.y, fmaf(w.z, v.z, fmaf(w.w, v.w, acc))));
    }
    for (int off = 16; off; off >>= 1) acc += __shfl_xor(acc, off, 32);
    if (lane == 0) {
        float r = acc + b[gw];
        y[gw] = relu ? fmaxf(r, 0.f) : r;
    }
}

// ---------------------------------------------------------------------------
// K3: softmax over 512 scores; result to ws and to d_out tail (attn_weights)
// One block of 512 threads (16 waves).
// ---------------------------------------------------------------------------
__global__ void k_softmax512(float* __restrict__ s,
                             float* __restrict__ out_tail) {
    __shared__ float red[16];
    __shared__ float stat[2];
    int t = threadIdx.x;
    float v = s[t];

    float m = v;
    for (int off = 16; off; off >>= 1) m = fmaxf(m, __shfl_xor(m, off, 32));
    if ((t & 31) == 0) red[t >> 5] = m;
    __syncthreads();
    if (t == 0) {
        float mm = red[0];
        for (int i = 1; i < 16; ++i) mm = fmaxf(mm, red[i]);
        stat[0] = mm;
    }
    __syncthreads();

    float e = __expf(v - stat[0]);
    float ss = e;
    for (int off = 16; off; off >>= 1) ss += __shfl_xor(ss, off, 32);
    if ((t & 31) == 0) red[t >> 5] = ss;
    __syncthreads();
    if (t == 0) {
        float tot = 0.f;
        for (int i = 0; i < 16; ++i) tot += red[i];
        stat[1] = tot;
    }
    __syncthreads();

    float w = e / stat[1];
    s[t] = w;
    out_tail[t] = w;
}

// ---------------------------------------------------------------------------
// K4: attn_applied[h] = sum_l w[l] * enc[l*H + h]   (coalesced over h)
// ---------------------------------------------------------------------------
__global__ void k_attn_apply(const float* __restrict__ w,
                             const float* __restrict__ enc,
                             float* __restrict__ out) {
    int h = blockIdx.x * blockDim.x + threadIdx.x;
    if (h >= H_DIM) return;
    float acc = 0.f;
    for (int l = 0; l < L_DIM; ++l)
        acc = fmaf(w[l], enc[(size_t)l * H_DIM + h], acc);
    out[h] = acc;
}

// ---------------------------------------------------------------------------
// K7: GRU elementwise combine (PyTorch r,z,n ordering)
// ---------------------------------------------------------------------------
__global__ void k_gru_combine(const float* __restrict__ gx,
                              const float* __restrict__ gh,
                              const float* __restrict__ h0,
                              float* __restrict__ hn) {
    int i = blockIdx.x * blockDim.x + threadIdx.x;
    if (i >= H_DIM) return;
    float r = 1.f / (1.f + __expf(-(gx[i]           + gh[i])));
    float z = 1.f / (1.f + __expf(-(gx[H_DIM + i]   + gh[H_DIM + i])));
    float n = tanhf(gx[2 * H_DIM + i] + r * gh[2 * H_DIM + i]);
    hn[i] = (1.f - z) * n + z * h0[i];
}

// ---------------------------------------------------------------------------
// K8: logits = h_new @ out_W.T + out_b   via v_wmma_f32_16x16x32_f16.
// Each wave owns 16 rows of out_W; lane l streams row row0+(l&15) contiguously.
// A fragment follows the ISA 16-bit A 16x32 layout (lanes>=16 take K+8 half).
// B fragment = h_new chunk broadcast to all 16 columns (same khalf packing).
// K unrolled x2 with two independent accumulators to break the WMMA C->D
// dependency chain and double in-flight global loads.
// All control flow around the WMMA loop is wave-uniform (EXEC stays all-1s).
// ---------------------------------------------------------------------------
__global__ void k_out_proj_wmma(const float* __restrict__ outW,
                                const float* __restrict__ outb,
                                const float* __restrict__ hn,
                                float* __restrict__ logits) {
    __shared__ _Float16 hs[H_DIM];
    for (int i = threadIdx.x; i < H_DIM; i += blockDim.x)
        hs[i] = (_Float16)hn[i];
    __syncthreads();

    int wave = threadIdx.x >> 5;          // 0..7
    int lane = threadIdx.x & 31;
    int row0 = (blockIdx.x * 8 + wave) * 16;
    if (row0 >= V_DIM) return;            // wave-uniform exit

    int  lrow  = lane & 15;
    int  khalf = (lane >> 4) ? 8 : 0;
    size_t r = (size_t)(row0 + lrow);
    if (r >= V_DIM) r = V_DIM - 1;        // clamp tail rows (EXEC stays full)
    const float* wrow = outW + r * H_DIM;

    v8f c0 = {};
    v8f c1 = {};
    for (int k0 = 0; k0 < H_DIM; k0 += 64) {
        v16h a0, b0, a1, b1;
#pragma unroll
        for (int i = 0; i < 8; ++i) {
            a0[i]     = (_Float16)wrow[k0 + khalf + i];
            a0[i + 8] = (_Float16)wrow[k0 + 16 + khalf + i];
            a1[i]     = (_Float16)wrow[k0 + 32 + khalf + i];
            a1[i + 8] = (_Float16)wrow[k0 + 48 + khalf + i];
            b0[i]     = hs[k0 + khalf + i];
            b0[i + 8] = hs[k0 + 16 + khalf + i];
            b1[i]     = hs[k0 + 32 + khalf + i];
            b1[i + 8] = hs[k0 + 48 + khalf + i];
        }
        c0 = __builtin_amdgcn_wmma_f32_16x16x32_f16(
                 false, a0, false, b0, (short)0, c0, false, false);
        c1 = __builtin_amdgcn_wmma_f32_16x16x32_f16(
                 false, a1, false, b1, (short)0, c1, false, false);
    }
    v8f c = c0 + c1;

    // D: VGPR j, lanes 0-15 -> M=j ; lanes 16-31 -> M=8+j. All N columns equal.
    if (row0 + 16 <= V_DIM) {
        // fast path: unguarded vectorized stores from lanes 0 and 16
        if (lane == 0 || lane == 16) {
            int base = row0 + ((lane == 16) ? 8 : 0);
            const float4* b4 = (const float4*)(outb + base);
            float4 r0 = b4[0];
            float4 r1 = b4[1];
            r0.x += c[0]; r0.y += c[1]; r0.z += c[2]; r0.w += c[3];
            r1.x += c[4]; r1.y += c[5]; r1.z += c[6]; r1.w += c[7];
            float4* o4 = (float4*)(logits + base);
            o4[0] = r0;
            o4[1] = r1;
        }
    } else {
        // tail block (one per launch): guarded scalar stores
        if (lane == 0) {
#pragma unroll
            for (int j = 0; j < 8; ++j) {
                int rr = row0 + j;
                if (rr < V_DIM) logits[rr] = c[j] + outb[rr];
            }
        } else if (lane == 16) {
#pragma unroll
            for (int j = 0; j < 8; ++j) {
                int rr = row0 + 8 + j;
                if (rr < V_DIM) logits[rr] = c[j] + outb[rr];
            }
        }
    }
}

// ---------------------------------------------------------------------------
// K9: single-block max + log-sum-exp over V logits -> red[0]=max, red[1]=log(sum)
// ---------------------------------------------------------------------------
__global__ void k_lse_reduce(const float* __restrict__ logits,
                             float* __restrict__ red) {
    __shared__ float rbuf[32];
    __shared__ float stat;
    int t = threadIdx.x;  // 1024 threads = 32 waves

    float m = -3.402823466e+38f;
    for (int i = t; i < V_DIM; i += 1024) m = fmaxf(m, logits[i]);
    for (int off = 16; off; off >>= 1) m = fmaxf(m, __shfl_xor(m, off, 32));
    if ((t & 31) == 0) rbuf[t >> 5] = m;
    __syncthreads();
    if (t == 0) {
        float mm = rbuf[0];
        for (int i = 1; i < 32; ++i) mm = fmaxf(mm, rbuf[i]);
        stat = mm;
    }
    __syncthreads();
    float M = stat;

    float s = 0.f;
    for (int i = t; i < V_DIM; i += 1024) s += __expf(logits[i] - M);
    for (int off = 16; off; off >>= 1) s += __shfl_xor(s, off, 32);
    if ((t & 31) == 0) rbuf[t >> 5] = s;
    __syncthreads();
    if (t == 0) {
        float tot = 0.f;
        for (int i = 0; i < 32; ++i) tot += rbuf[i];
        red[0] = M;
        red[1] = logf(tot);
    }
}

// ---------------------------------------------------------------------------
// K10: logp[v] = logit[v] - max - log(sum)   (in place in d_out)
// ---------------------------------------------------------------------------
__global__ void k_logp_final(float* __restrict__ logits,
                             const float* __restrict__ red) {
    int i = blockIdx.x * blockDim.x + threadIdx.x;
    if (i < V_DIM) logits[i] = logits[i] - red[0] - red[1];
}

// ---------------------------------------------------------------------------
// Host launcher
// ---------------------------------------------------------------------------
extern "C" void kernel_launch(void* const* d_in, const int* in_sizes, int n_in,
                              void* d_out, int out_size, void* d_ws, size_t ws_size,
                              hipStream_t stream) {
    const int*   x      = (const int*)  d_in[0];
    const float* enc    = (const float*)d_in[1];
    const float* hidden = (const float*)d_in[2];   // h0, H floats
    const float* emb    = (const float*)d_in[3];
    const float* attnW  = (const float*)d_in[4];
    const float* attnB  = (const float*)d_in[5];
    const float* combW  = (const float*)d_in[6];
    const float* combB  = (const float*)d_in[7];
    const float* W_ih   = (const float*)d_in[8];
    const float* W_hh   = (const float*)d_in[9];
    const float* b_ih   = (const float*)d_in[10];
    const float* b_hh   = (const float*)d_in[11];
    const float* outW   = (const float*)d_in[12];
    const float* outB   = (const float*)d_in[13];

    float* out = (float*)d_out;                 // [logp(V) | attn_weights(L)]
    float* ws  = (float*)d_ws;

    // workspace layout (floats); all offsets 16B-aligned
    float* XCAT = ws + 0;        // 2048: [embedded | h0] -> later [embedded | attn_applied]
    float* S    = ws + 2048;     // 512 : attn scores -> softmax weights
    float* GI   = ws + 2560;     // 1024: GRU input (relu combine)
    float* GX   = ws + 3584;     // 3072: input gates
    float* GH   = ws + 6656;     // 3072: hidden gates
    float* HN   = ws + 9728;     // 1024: h_new
    float* RED  = ws + 10752;    // 2   : max, log(sumexp)

    // 1. xcat = [emb[x] | h0]
    k_build_xcat<<<(2 * H_DIM + 255) / 256, 256, 0, stream>>>(x, emb, hidden, XCAT);

    // 2. attn scores: S = attn_W(512x2048) @ xcat + attn_b
    k_gemv_rows<<<(L_DIM * 32 + 255) / 256, 256, 0, stream>>>(
        attnW, XCAT, attnB, S, L_DIM, 2 * H_DIM, 0);

    // 3. softmax -> S and attn_weights output
    k_softmax512<<<1, 512, 0, stream>>>(S, out + V_DIM);

    // 4. attn_applied = S @ enc  -> overwrite second half of XCAT
    k_attn_apply<<<(H_DIM + 255) / 256, 256, 0, stream>>>(S, enc, XCAT + H_DIM);

    // 5. GRU input = relu(comb_W(1024x2048) @ [embedded|attn_applied] + comb_b)
    k_gemv_rows<<<(H_DIM * 32 + 255) / 256, 256, 0, stream>>>(
        combW, XCAT, combB, GI, H_DIM, 2 * H_DIM, 1);

    // 6. gate GEMVs: GX = W_ih @ GI + b_ih ; GH = W_hh @ h0 + b_hh
    k_gemv_rows<<<(3 * H_DIM * 32 + 255) / 256, 256, 0, stream>>>(
        W_ih, GI, b_ih, GX, 3 * H_DIM, H_DIM, 0);
    k_gemv_rows<<<(3 * H_DIM * 32 + 255) / 256, 256, 0, stream>>>(
        W_hh, hidden, b_hh, GH, 3 * H_DIM, H_DIM, 0);

    // 7. h_new
    k_gru_combine<<<(H_DIM + 255) / 256, 256, 0, stream>>>(GX, GH, hidden, HN);

    // 8. logits = h_new @ out_W.T + out_b  (WMMA; 128 rows per block)
    int proj_blocks = (V_DIM + 127) / 128;     // 393
    k_out_proj_wmma<<<proj_blocks, 256, 0, stream>>>(outW, outB, HN, out);

    // 9+10. log_softmax in place
    k_lse_reduce<<<1, 1024, 0, stream>>>(out, RED);
    k_logp_final<<<(V_DIM + 255) / 256, 256, 0, stream>>>(out, RED);
}
